// FlashAttention_12592844112132
// MI455X (gfx1250) — matrix operand order repaired and checked by reading, not compiled, not run on hardware
//
#include <hip/hip_runtime.h>
#include <hip/hip_bf16.h>
#include <math.h>

// ---------------------------------------------------------------------------
// Multi-head attention block (B=4, N=2048, C=1024, H=16, D=64) for gfx1250.
// All GEMMs run on v_wmma_f32_16x16x32_f16 (f16 in, f32 accumulate).
//   1) convert x -> f16, transpose+convert w_qkv / w_proj -> f16 (BT layout)
//   2) WMMA GEMM (64x64/wave): qkv = x @ w_qkv ; scatter to Q(*SCALE), K, V^T
//   3) flash attention (32 q-rows/wave): streaming softmax, WMMA QK^T and PV
//   4) WMMA GEMM (64x64/wave): out = attn @ w_proj + b_proj (f32 out)
// Working set (~100MB) is L2-resident (192MB); 64x64 register tiles give a
// 1:1 b128-load : wmma ratio (~512 B of L2 traffic per 16-KFLOP WMMA).
// ---------------------------------------------------------------------------

typedef __attribute__((ext_vector_type(16))) _Float16 v16h;
typedef __attribute__((ext_vector_type(8)))  float    v8f;
typedef __attribute__((ext_vector_type(4)))  float    v4f;

#define NB 4
#define NSEQ 2048
#define CDIM 1024
#define NH 16
#define HD 64
#define QKV_N 3072
#define MROWS (NB * NSEQ) /* 8192 */
#define ATTN_SCALE 0.125f /* 64^-0.5 */

union Frag16 { v16h h; v4f q[2]; };

// A operand (16x32 f16 tile, row-major source, leading dim ld in elems).
// wave32 layout: lanes 0-15 rows 0-15 with K = {8*kh..8*kh+7, 16+8*kh..}.
__device__ __forceinline__ v16h load_a_frag(const _Float16* __restrict__ base,
                                            int ld, int lane) {
  const int r  = lane & 15;
  const int kh = (lane >> 4) & 1;
  const _Float16* p = base + (size_t)r * ld + kh * 8;
  Frag16 f;
  f.q[0] = *(const v4f*)(p);
  f.q[1] = *(const v4f*)(p + 16);
  return f.h;
}

// B operand (32x16 f16 tile). bt points at BT tile start: BT[col][k],
// row-major with leading dim ld. Lanes 0-15 = cols, K = {8*kh..8*kh+7, 16+8*kh..}.
__device__ __forceinline__ v16h load_b_frag(const _Float16* __restrict__ bt,
                                            int ld, int lane) {
  const int c  = lane & 15;
  const int kh = (lane >> 4) & 1;
  const _Float16* p = bt + (size_t)c * ld + kh * 8;
  Frag16 f;
  f.q[0] = *(const v4f*)(p);
  f.q[1] = *(const v4f*)(p + 16);
  return f.h;
}

__device__ __forceinline__ v8f wmma16(v16h a, v16h b, v8f c) {
  return __builtin_amdgcn_wmma_f32_16x16x32_f16(false, a, false, b,
                                                (short)0, c, false, false);
}

// ------------------------- conversion kernels ------------------------------

__global__ __launch_bounds__(256) void cvt_f32_to_f16(
    const float* __restrict__ src, _Float16* __restrict__ dst, int n) {
  int i = blockIdx.x * 256 + threadIdx.x;
  if (i < n) dst[i] = (_Float16)src[i];
}

// dst[n][k] = (f16) src[k][n]   (src is rows x cols, dst is cols x rows)
__global__ __launch_bounds__(256) void transpose_cvt(
    const float* __restrict__ src, _Float16* __restrict__ dst,
    int rows, int cols) {
  int i = blockIdx.x * 256 + threadIdx.x;
  if (i < rows * cols) {
    int n = i / rows;
    int k = i - n * rows;
    dst[i] = (_Float16)src[(size_t)k * cols + n];
  }
}

// ------------------------- GEMM 1: qkv projection --------------------------
// C[8192 x 3072] = x16[8192 x 1024] @ w_qkvT(BT)[3072 x 1024]
// 256 threads = 8 waves (2x4), 64x64 per wave -> 128x256 block tile.
__global__ __launch_bounds__(256) void gemm_qkv(
    const _Float16* __restrict__ x16, const _Float16* __restrict__ wT,
    _Float16* __restrict__ Q, _Float16* __restrict__ K,
    _Float16* __restrict__ VT) {
  const int lane = threadIdx.x & 31;
  const int wave = threadIdx.x >> 5;
  const int ntiles = QKV_N / 256; // 12
  const int bm = blockIdx.x / ntiles;
  const int bn = blockIdx.x - bm * ntiles;
  const int m0 = bm * 128 + (wave >> 2) * 64;
  const int n0 = bn * 256 + (wave & 3) * 64;

  v8f acc[4][4] = {};
  const _Float16* abase = x16 + (size_t)m0 * CDIM;
  const _Float16* bbase = wT + (size_t)n0 * CDIM;

  for (int k = 0; k < CDIM; k += 32) {
    v16h a[4], b[4];
#pragma unroll
    for (int i = 0; i < 4; ++i)
      a[i] = load_a_frag(abase + (size_t)(i * 16) * CDIM + k, CDIM, lane);
#pragma unroll
    for (int j = 0; j < 4; ++j)
      b[j] = load_b_frag(bbase + (size_t)(j * 16) * CDIM + k, CDIM, lane);
#pragma unroll
    for (int i = 0; i < 4; ++i)
#pragma unroll
      for (int j = 0; j < 4; ++j)
        acc[i][j] = wmma16(a[i], b[j], acc[i][j]);
  }

  // Epilogue: scatter column c -> (t, h, d); t=0:Q(*SCALE) t=1:K t=2:V^T.
  const int half = lane >> 4;
  const int cl   = lane & 15;
#pragma unroll
  for (int mi = 0; mi < 4; ++mi) {
#pragma unroll
    for (int ni = 0; ni < 4; ++ni) {
      const int col = n0 + ni * 16 + cl;
      const int t = col >> 10;          // which of q/k/v
      const int h = (col & 1023) >> 6;  // head
      const int d = col & 63;           // dim within head
#pragma unroll
      for (int i = 0; i < 8; ++i) {
        const int m  = m0 + mi * 16 + half * 8 + i;
        const int b  = m >> 11;
        const int nr = m & (NSEQ - 1);
        const size_t bh = (size_t)b * NH + h;
        const float v = acc[mi][ni][i];
        if (t == 0)
          Q[(bh * NSEQ + nr) * HD + d] = (_Float16)(v * ATTN_SCALE);
        else if (t == 1)
          K[(bh * NSEQ + nr) * HD + d] = (_Float16)v;
        else
          VT[(bh * HD + d) * NSEQ + nr] = (_Float16)v;
      }
    }
  }
}

// ------------------------- Kernel 2: flash attention -----------------------
// grid = (B*H) * (N/128) blocks, 128 threads (4 waves), 32 q-rows/wave
// (two 16-row tiles sharing every K/V fragment load).
__global__ __launch_bounds__(128) void flash_attn(
    const _Float16* __restrict__ Q, const _Float16* __restrict__ K,
    const _Float16* __restrict__ VT, _Float16* __restrict__ O16) {
  __shared__ _Float16 plds[4][2 * 16 * 64];

  const int lane = threadIdx.x & 31;
  const int wave = threadIdx.x >> 5;
  const int nqt  = NSEQ / 128;                // 16
  const int qt   = blockIdx.x % nqt;
  const int bh   = blockIdx.x / nqt;          // 0..63
  const int b    = bh >> 4;
  const int h    = bh & 15;

  const _Float16* Qb = Q  + (size_t)bh * NSEQ * HD;
  const _Float16* Kb = K  + (size_t)bh * NSEQ * HD;
  const _Float16* Vb = VT + (size_t)bh * HD * NSEQ;

  const int qrow = qt * 128 + wave * 32;      // first of 32 rows for this wave
  // Q fragments (2 row-tiles x 2 k-halves), SCALE folded in, kept resident.
  v16h qf[2][2];
#pragma unroll
  for (int r = 0; r < 2; ++r) {
    qf[r][0] = load_a_frag(Qb + (size_t)(qrow + r * 16) * HD + 0,  HD, lane);
    qf[r][1] = load_a_frag(Qb + (size_t)(qrow + r * 16) * HD + 32, HD, lane);
  }

  v8f o[2][4] = {};
  float mrow[2][8], lrow[2][8];
#pragma unroll
  for (int r = 0; r < 2; ++r)
#pragma unroll
    for (int i = 0; i < 8; ++i) { mrow[r][i] = -__builtin_inff(); lrow[r][i] = 0.f; }

  _Float16* myp = plds[wave];
  const int half = lane >> 4;
  const int cl   = lane & 15;

  for (int kv = 0; kv < NSEQ; kv += 64) {
    // ---- S = Q @ K^T, 32x64 tile; each K fragment feeds both row tiles ----
    v8f s[2][4];
#pragma unroll
    for (int c = 0; c < 4; ++c) {
      const _Float16* kb = Kb + (size_t)(kv + c * 16) * HD;
      v16h kb0 = load_b_frag(kb + 0,  HD, lane);
      v16h kb1 = load_b_frag(kb + 32, HD, lane);
#pragma unroll
      for (int r = 0; r < 2; ++r) {
        v8f z = {};
        s[r][c] = wmma16(qf[r][0], kb0, z);
        s[r][c] = wmma16(qf[r][1], kb1, s[r][c]);
      }
    }

    // ---- streaming softmax per row tile; stage P into LDS ----
#pragma unroll
    for (int r = 0; r < 2; ++r) {
      float mnew[8], corr[8];
#pragma unroll
      for (int i = 0; i < 8; ++i) {
        float mx = fmaxf(fmaxf(s[r][0][i], s[r][1][i]),
                         fmaxf(s[r][2][i], s[r][3][i]));
#pragma unroll
        for (int off = 8; off >= 1; off >>= 1)
          mx = fmaxf(mx, __shfl_xor(mx, off, 32)); // stays in 16-lane half
        mnew[i] = fmaxf(mrow[r][i], mx);
        corr[i] = __expf(mrow[r][i] - mnew[i]);
        mrow[r][i] = mnew[i];
      }
#pragma unroll
      for (int c = 0; c < 4; ++c)
#pragma unroll
        for (int i = 0; i < 8; ++i)
          s[r][c][i] = __expf(s[r][c][i] - mnew[i]);
#pragma unroll
      for (int i = 0; i < 8; ++i) {
        float sm = s[r][0][i] + s[r][1][i] + s[r][2][i] + s[r][3][i];
#pragma unroll
        for (int off = 8; off >= 1; off >>= 1)
          sm += __shfl_xor(sm, off, 32);
        lrow[r][i] = lrow[r][i] * corr[i] + sm;
      }
#pragma unroll
      for (int c = 0; c < 4; ++c)
#pragma unroll
        for (int i = 0; i < 8; ++i)
          o[r][c][i] *= corr[i];
      // C-layout -> row-major 16x64 in per-wave LDS region
#pragma unroll
      for (int c = 0; c < 4; ++c)
#pragma unroll
        for (int i = 0; i < 8; ++i)
          myp[r * 1024 + (half * 8 + i) * 64 + c * 16 + cl] =
              (_Float16)s[r][c][i];
    }
    __syncthreads();

    // ---- O += P @ V; each V fragment feeds both row tiles ----
    v16h pf[2][2];
#pragma unroll
    for (int r = 0; r < 2; ++r) {
      pf[r][0] = load_a_frag(myp + r * 1024 + 0,  64, lane);
      pf[r][1] = load_a_frag(myp + r * 1024 + 32, 64, lane);
    }
#pragma unroll
    for (int c = 0; c < 4; ++c) {
      const _Float16* vb = Vb + (size_t)(c * 16) * NSEQ + kv;
      v16h vb0 = load_b_frag(vb + 0,  NSEQ, lane);
      v16h vb1 = load_b_frag(vb + 32, NSEQ, lane);
#pragma unroll
      for (int r = 0; r < 2; ++r) {
        o[r][c] = wmma16(pf[r][0], vb0, o[r][c]);
        o[r][c] = wmma16(pf[r][1], vb1, o[r][c]);
      }
    }
    __syncthreads();
  }

  // ---- epilogue: divide by l, store (B,N,C) f16 ----
#pragma unroll
  for (int r = 0; r < 2; ++r) {
#pragma unroll
    for (int i = 0; i < 8; ++i) lrow[r][i] = 1.0f / lrow[r][i];
#pragma unroll
    for (int c = 0; c < 4; ++c)
#pragma unroll
      for (int i = 0; i < 8; ++i) {
        const int rr = qrow + r * 16 + half * 8 + i;
        O16[((size_t)b * NSEQ + rr) * CDIM + h * HD + c * 16 + cl] =
            (_Float16)(o[r][c][i] * lrow[r][i]);
      }
  }
}

// ------------------------- GEMM 2: output projection -----------------------
__global__ __launch_bounds__(256) void gemm_proj(
    const _Float16* __restrict__ a16, const _Float16* __restrict__ wT,
    const float* __restrict__ bias, float* __restrict__ out) {
  const int lane = threadIdx.x & 31;
  const int wave = threadIdx.x >> 5;
  const int ntiles = CDIM / 256; // 4
  const int bm = blockIdx.x / ntiles;
  const int bn = blockIdx.x - bm * ntiles;
  const int m0 = bm * 128 + (wave >> 2) * 64;
  const int n0 = bn * 256 + (wave & 3) * 64;

  v8f acc[4][4] = {};
  const _Float16* abase = a16 + (size_t)m0 * CDIM;
  const _Float16* bbase = wT + (size_t)n0 * CDIM;

  for (int k = 0; k < CDIM; k += 32) {
    v16h a[4], b[4];
#pragma unroll
    for (int i = 0; i < 4; ++i)
      a[i] = load_a_frag(abase + (size_t)(i * 16) * CDIM + k, CDIM, lane);
#pragma unroll
    for (int j = 0; j < 4; ++j)
      b[j] = load_b_frag(bbase + (size_t)(j * 16) * CDIM + k, CDIM, lane);
#pragma unroll
    for (int i = 0; i < 4; ++i)
#pragma unroll
      for (int j = 0; j < 4; ++j)
        acc[i][j] = wmma16(a[i], b[j], acc[i][j]);
  }

  const int half = lane >> 4;
  const int cl   = lane & 15;
#pragma unroll
  for (int mi = 0; mi < 4; ++mi) {
#pragma unroll
    for (int ni = 0; ni < 4; ++ni) {
      const int col = n0 + ni * 16 + cl;
      const float bb = bias[col];
#pragma unroll
      for (int i = 0; i < 8; ++i) {
        const int m = m0 + mi * 16 + half * 8 + i;
        out[(size_t)m * CDIM + col] = acc[mi][ni][i] + bb;
      }
    }
  }
}

// --------------------------------- launch ----------------------------------

extern "C" void kernel_launch(void* const* d_in, const int* in_sizes, int n_in,
                              void* d_out, int out_size, void* d_ws, size_t ws_size,
                              hipStream_t stream) {
  const float* x      = (const float*)d_in[0]; // (4,2048,1024)
  const float* w_qkv  = (const float*)d_in[1]; // (1024,3072)
  const float* w_proj = (const float*)d_in[2]; // (1024,1024)
  const float* b_proj = (const float*)d_in[3]; // (1024)
  float* out = (float*)d_out;                  // (4,2048,1024)

  // Workspace layout (f16), total ~92 MB.
  char* ws = (char*)d_ws;
  const size_t xN   = (size_t)MROWS * CDIM;        // 8388608
  const size_t qkvN = (size_t)CDIM * QKV_N;        // 3145728
  const size_t pN   = (size_t)CDIM * CDIM;         // 1048576
  const size_t hN   = (size_t)NB * NH * NSEQ * HD; // 8388608

  _Float16* x16    = (_Float16*)(ws);
  _Float16* wqkvT  = (_Float16*)(ws + 2 * xN);
  _Float16* wprojT = (_Float16*)(ws + 2 * (xN + qkvN));
  _Float16* Qd     = (_Float16*)(ws + 2 * (xN + qkvN + pN));
  _Float16* Kd     = (_Float16*)(ws + 2 * (xN + qkvN + pN + hN));
  _Float16* VTd    = (_Float16*)(ws + 2 * (xN + qkvN + pN + 2 * hN));
  _Float16* attn16 = (_Float16*)(ws + 2 * (xN + qkvN + pN + 3 * hN));

  // 1) precision conversion + weight transposes
  cvt_f32_to_f16<<<(int)((xN + 255) / 256), 256, 0, stream>>>(x, x16, (int)xN);
  transpose_cvt<<<(int)((qkvN + 255) / 256), 256, 0, stream>>>(w_qkv, wqkvT,
                                                               CDIM, QKV_N);
  transpose_cvt<<<(int)((pN + 255) / 256), 256, 0, stream>>>(w_proj, wprojT,
                                                             CDIM, CDIM);

  // 2) qkv projection (WMMA) with Q/K/V^T scatter epilogue
  gemm_qkv<<<(MROWS / 128) * (QKV_N / 256), 256, 0, stream>>>(x16, wqkvT,
                                                              Qd, Kd, VTd);

  // 3) flash attention (WMMA + streaming softmax)
  flash_attn<<<NB * NH * (NSEQ / 128), 128, 0, stream>>>(Qd, Kd, VTd, attn16);

  // 4) output projection (WMMA) + bias, f32 output
  gemm_proj<<<(MROWS / 128) * (CDIM / 256), 256, 0, stream>>>(attn16, wprojT,
                                                              b_proj, out);
}
